// BConv_case1_45835890983450
// MI455X (gfx1250) — matrix operand order repaired and checked by reading, not compiled, not run on hardware
//
#include <hip/hip_runtime.h>

// ---------------------------------------------------------------------------
// Binary-weight 3x3 conv (256->256, 56x56, N=32, pad=1) as WMMA implicit GEMM
//   M = 256 (out channels), N = 100352 (pixels), K = 2304 ((kh,kw)-major, ic)
// bf16 WMMA: sign(w) in {-1,0,+1} is exact in bf16; activations fp32->bf16.
// A-tile moved with gfx1250 async global->LDS; B-tile is a packed im2col
// gather (2x ds_store_b128 per thread per K-step).
// ---------------------------------------------------------------------------

typedef __attribute__((ext_vector_type(16))) __bf16 v16bf;
typedef __attribute__((ext_vector_type(8)))  __bf16 v8bf;
typedef __attribute__((ext_vector_type(8)))  float  v8f;

#define OUT_CHN 256
#define IN_CHN  256
#define HW      56
#define IMG     (HW * HW)          // 3136
#define NBATCH  32
#define NPIX    (NBATCH * IMG)     // 100352
#define KCH     (IN_CHN * 9)       // 2304
#define TILE_M  128
#define TILE_N  128
#define LDSS    40                 // bf16 elements per LDS row (80 B, skewed)

// --- prep: binarize fp32 weights and reorder K as (kh*3+kw)*256 + ic --------
__global__ void binw_kernel(const float* __restrict__ w, __bf16* __restrict__ wb) {
  int idx = blockIdx.x * 256 + threadIdx.x;          // over 256*2304
  if (idx >= OUT_CHN * KCH) return;
  int oc   = idx / KCH;
  int r    = idx - oc * KCH;
  int ic   = r / 9;
  int kpos = r - ic * 9;                             // kh*3+kw
  float v  = w[idx];
  float s  = (v > 0.0f) ? 1.0f : ((v < 0.0f) ? -1.0f : 0.0f);
  wb[oc * KCH + kpos * IN_CHN + ic] = (__bf16)s;
}

// --- main: 128x128 output tile per block, 8 wave32s, 2x4 WMMA per wave ------
__global__ __launch_bounds__(256)
void bconv_wmma_kernel(const float* __restrict__ x,
                       const __bf16* __restrict__ wb,
                       const float* __restrict__ scale,
                       float* __restrict__ out) {
  __shared__ __align__(16) __bf16 Alds[TILE_M * LDSS];   // [oc_local][k]
  __shared__ __align__(16) __bf16 Blds[TILE_N * LDSS];   // [px_local][k]
  __shared__ float Sc[TILE_M];                           // per-oc scales

  const int tid  = threadIdx.x;
  const int lane = tid & 31;
  const int wid  = tid >> 5;
  const int wm   = wid & 3;                 // 4 waves along M (32 oc each)
  const int wn   = wid >> 2;                // 2 waves along N (64 px each)
  const int ocb  = blockIdx.y * TILE_M;
  const int pxb  = blockIdx.x * TILE_N;

  if (tid < TILE_M) Sc[tid] = scale[ocb + tid];   // covered by first barrier

  v8f acc[2][4];
#pragma unroll
  for (int tm = 0; tm < 2; ++tm)
#pragma unroll
    for (int tn = 0; tn < 4; ++tn) acc[tm][tn] = (v8f){0,0,0,0,0,0,0,0};

  // B writer mapping: this thread owns pixel pxl and 16 consecutive ic.
  const int pxl = tid & (TILE_N - 1);       // 0..127
  const int icg = tid >> 7;                 // 0..1 -> ic sub-block of 16
  int px  = pxb + pxl;
  int n   = px / IMG;
  int rem = px - n * IMG;
  int ph  = rem / HW;
  int pw  = rem - ph * HW;
  const int nbase = (n * IN_CHN + icg * 16) * IMG;   // + ic*IMG + ih*56 + iw
  // LDS row segment this thread fills: Blds[pxl][icg*16 .. icg*16+15]
  __bf16* bdst = &Blds[pxl * LDSS + icg * 16];

  // A async-copy mapping: thread moves 32 B of one 64 B weight row.
  const int arow = tid >> 1;                // 0..127
  const int aseg = tid & 1;                 // 0..1
  const unsigned lds_a =
      (unsigned)(uintptr_t)&Alds[arow * LDSS + aseg * 16];  // LDS byte offset

  for (int kpos = 0; kpos < 9; ++kpos) {
    const int dh = kpos / 3 - 1;
    const int dw = kpos % 3 - 1;
    const int ih = ph + dh;
    const int iw = pw + dw;
    const bool ok = ((unsigned)ih < (unsigned)HW) && ((unsigned)iw < (unsigned)HW);
    const int hwoff = ih * HW + iw;

#pragma unroll 2
    for (int icb8 = 0; icb8 < 8; ++icb8) {
      const int icb = icb8 * 32;
      __syncthreads();                      // previous tile fully consumed

      // A tile: 128 oc x 32 k bf16 weights, async global -> LDS (b128 x2)
      {
        const __bf16* src =
            wb + (ocb + arow) * KCH + kpos * IN_CHN + icb + aseg * 16;
        unsigned long long ga = (unsigned long long)(uintptr_t)src;
        asm volatile(
            "global_load_async_to_lds_b128 %0, %1, off\n\t"
            "global_load_async_to_lds_b128 %2, %3, off"
            :
            : "v"(lds_a), "v"(ga), "v"(lds_a + 16u), "v"(ga + 16ull)
            : "memory");
        __builtin_prefetch(src + 32, 0, 1);               // next ic-block tile
      }

      // B tile: im2col gather, 16 consecutive ic for one pixel; pad -> 0.
      {
        float v[16];
        if (ok) {
          const float* gsrc = x + nbase + icb * IMG + hwoff;
#pragma unroll
          for (int j = 0; j < 16; ++j) v[j] = gsrc[j * IMG];  // imm offsets
        } else {
#pragma unroll
          for (int j = 0; j < 16; ++j) v[j] = 0.0f;
        }
        v8bf p0, p1;
#pragma unroll
        for (int j = 0; j < 8; ++j) { p0[j] = (__bf16)v[j]; p1[j] = (__bf16)v[j + 8]; }
        *(v8bf*)bdst       = p0;            // ds_store_b128
        *(v8bf*)(bdst + 8) = p1;            // ds_store_b128
      }

      asm volatile("s_wait_asynccnt 0x0" ::: "memory");   // A tile landed
      __syncthreads();                      // tiles visible to all waves

      // Fragment loads: lanes<16 hold K 0..15, lanes>=16 hold K 16..31
      // (A and B both group lanes by contiguous K halves).
      const int aoff = (lane < 16) ? 0 : 16;
      const int boff = (lane < 16) ? 0 : 16;
      v16bf afrag[2], bfrag[4];
#pragma unroll
      for (int tm = 0; tm < 2; ++tm) {
        const __bf16* p = &Alds[(wm * 32 + tm * 16 + (lane & 15)) * LDSS + aoff];
        v8bf lo = *(const v8bf*)p;
        v8bf hi = *(const v8bf*)(p + 8);
        afrag[tm] = __builtin_shufflevector(lo, hi, 0,1,2,3,4,5,6,7,8,9,10,11,12,13,14,15);
      }
#pragma unroll
      for (int tn = 0; tn < 4; ++tn) {
        const __bf16* p = &Blds[(wn * 64 + tn * 16 + (lane & 15)) * LDSS + boff];
        v8bf lo = *(const v8bf*)p;
        v8bf hi = *(const v8bf*)(p + 8);
        bfrag[tn] = __builtin_shufflevector(lo, hi, 0,1,2,3,4,5,6,7,8,9,10,11,12,13,14,15);
      }

#pragma unroll
      for (int tm = 0; tm < 2; ++tm)
#pragma unroll
        for (int tn = 0; tn < 4; ++tn)
          acc[tm][tn] = __builtin_amdgcn_wmma_f32_16x16x32_bf16(
              /*neg_a=*/false, afrag[tm], /*neg_b=*/false, bfrag[tn],
              /*c_mod=*/(short)0, acc[tm][tn], /*reuse_a=*/false, /*reuse_b=*/false);
    }
  }

  // Epilogue: per-output-channel scale (from LDS), scatter to NCHW fp32.
  // C/D layout: VGPR r, lanes 0-15 -> M=r, lanes 16-31 -> M=8+r; N = lane%16.
#pragma unroll
  for (int tm = 0; tm < 2; ++tm) {
#pragma unroll
    for (int tn = 0; tn < 4; ++tn) {
      int pcol = pxb + wn * 64 + tn * 16 + (lane & 15);
      int on   = pcol / IMG;
      int orem = pcol - on * IMG;
      float* obase = out + (size_t)(on * OUT_CHN) * IMG + orem;
      int ocl0 = wm * 32 + tm * 16 + ((lane >> 4) << 3);  // oc - ocb
#pragma unroll
      for (int r = 0; r < 8; ++r) {
        int ocl = ocl0 + r;
        obase[(size_t)(ocb + ocl) * IMG] = acc[tm][tn][r] * Sc[ocl];
      }
    }
  }
}

extern "C" void kernel_launch(void* const* d_in, const int* in_sizes, int n_in,
                              void* d_out, int out_size, void* d_ws, size_t ws_size,
                              hipStream_t stream) {
  const float* x   = (const float*)d_in[0];   // (32,256,56,56) fp32
  const float* w   = (const float*)d_in[1];   // (256*256*9, 1) fp32
  const float* sf  = (const float*)d_in[2];   // (1,256,1,1)   fp32
  float*       out = (float*)d_out;           // (32,256,56,56) fp32
  __bf16*      wb  = (__bf16*)d_ws;           // 256*2304 bf16 = 1.125 MB

  // 1) binarize + reorder weights
  int welems = OUT_CHN * KCH;                 // 589824
  binw_kernel<<<(welems + 255) / 256, 256, 0, stream>>>(w, wb);

  // 2) implicit-GEMM WMMA conv: grid = 784 pixel tiles x 2 oc tiles
  dim3 grid(NPIX / TILE_N, OUT_CHN / TILE_M); // (784, 2)
  bconv_wmma_kernel<<<grid, 256, 0, stream>>>(x, wb, sf, out);
}
